// Net_65369402245309
// MI455X (gfx1250) — matrix operand order repaired and checked by reading, not compiled, not run on hardware
//
#include <hip/hip_runtime.h>
#include <hip/hip_bf16.h>

// ---------------------------------------------------------------------------
// HMM forward (N=32,T=512,K=64,V=32000,D=100) fused for MI455X (gfx1250).
// - tran GEMM via v_wmma_f32_16x16x32_bf16; 16x4096 chunk lives only in LDS
// - EmT = (emb_cluster @ vocab^T)^T via WMMA, stored [V,64] for coalesced gather
// - one 512-thread workgroup per batch element; 8-way-parallel alpha scan with
//   ping-pong buffers (2 barriers/step)
// ---------------------------------------------------------------------------

#define HN 32
#define HT 512
#define HK 64
#define HV 32000
#define HD 100
#define DPAD 128
#define HE (HK*HK)          // 4096
#define TB 16               // timesteps per chunk
#define SCAN_G 8            // i-reduction groups in the scan step

typedef __attribute__((ext_vector_type(16))) __bf16 v16bf;
typedef __attribute__((ext_vector_type(8)))  float  v8f;

union V16 { v16bf v; unsigned short s[16]; uint4 q[2]; };

__device__ __forceinline__ unsigned short f2bf(float f) {
    union { float f; unsigned int u; } c; c.f = f;
    unsigned int u = c.u;
    return (unsigned short)((u + 0x7FFFu + ((u >> 16) & 1u)) >> 16);  // RNE
}
__device__ __forceinline__ float bf2f(unsigned short h) {
    union { unsigned int u; float f; } c; c.u = ((unsigned int)h) << 16;
    return c.f;
}

#define NEGINF (-3.402823466e38f)

// ---------------------------------------------------------------------------
// Kernel 1: trans_w [4096,100] f32 -> trans_bf [4096,128] bf16 (zero padded)
// ---------------------------------------------------------------------------
__global__ void k_prep_trans(const float* __restrict__ tw,
                             unsigned short* __restrict__ tb) {
    int idx = blockIdx.x * 256 + threadIdx.x;        // 4096*128 elements
    int e = idx >> 7, d = idx & 127;
    float v = (d < HD) ? tw[e * HD + d] : 0.0f;
    tb[idx] = f2bf(v);
}

// ---------------------------------------------------------------------------
// Kernel 2: EmT[v,k] = sum_c vocab[v,c] * embc[k,c]   (M=V, N=64, Kdim=64)
// 8 waves/block, each wave owns a 16-row M tile; 4 N-tiles x 2 k-steps = 8 WMMA
// ---------------------------------------------------------------------------
__global__ __launch_bounds__(256) void k_em(const float* __restrict__ embc,
                                            const float* __restrict__ vocab,
                                            float* __restrict__ EmT) {
    __shared__ unsigned short ec[HK * HK];           // embc in bf16, 8 KB
    for (int i = threadIdx.x; i < HK * HK; i += 256)
        ec[i] = f2bf(embc[i]);
    __syncthreads();

    const int lane = threadIdx.x & 31;
    const int hi   = lane >> 4;                      // upper half-wave
    const int mrow = lane & 15;
    const int wave = threadIdx.x >> 5;
    const int v0   = blockIdx.x * 128 + wave * 16;   // grid = 250 -> covers 32000

    // A fragments (vocab rows, converted inline): 2 k-steps of 32
    V16 afr[2];
    const float* vr = vocab + (size_t)(v0 + mrow) * HK;
#pragma unroll
    for (int kk = 0; kk < 2; ++kk) {
        int b0 = kk * 32 + hi * 8;
#pragma unroll
        for (int e = 0; e < 8; ++e) afr[kk].s[e]     = f2bf(vr[b0 + e]);
#pragma unroll
        for (int e = 0; e < 8; ++e) afr[kk].s[8 + e] = f2bf(vr[b0 + 16 + e]);
    }

#pragma unroll
    for (int nt = 0; nt < 4; ++nt) {                 // 4 tiles cover k = 0..63
        v8f acc = {0.f,0.f,0.f,0.f,0.f,0.f,0.f,0.f};
        const int kcol = nt * 16 + mrow;             // this lane's output column
#pragma unroll
        for (int kk = 0; kk < 2; ++kk) {
            V16 bu;
            const unsigned short* er = ec + kcol * HK + kk * 32 + hi * 8;
            bu.q[0] = *(const uint4*)(er);
            bu.q[1] = *(const uint4*)(er + 16);
            acc = __builtin_amdgcn_wmma_f32_16x16x32_bf16(
                false, afr[kk].v, false, bu.v, (short)0, acc, false, false);
        }
#pragma unroll
        for (int r = 0; r < 8; ++r) {
            int mm = r + hi * 8;                     // D layout: M = r (+8 for hi)
            EmT[(size_t)(v0 + mm) * HK + nt * 16 + mrow] = acc[r];
        }
    }
}

// ---------------------------------------------------------------------------
// Kernel 3: lse_k[k] = logsumexp_v EmT[v,k]   (one block per k)
// ---------------------------------------------------------------------------
__global__ __launch_bounds__(256) void k_lse(const float* __restrict__ EmT,
                                             float* __restrict__ lse_k) {
    const int k = blockIdx.x;
    __shared__ float red[256];
    float m = NEGINF;
    for (int v = threadIdx.x; v < HV; v += 256)
        m = fmaxf(m, EmT[(size_t)v * HK + k]);
    red[threadIdx.x] = m; __syncthreads();
    for (int s = 128; s > 0; s >>= 1) {
        if (threadIdx.x < s) red[threadIdx.x] = fmaxf(red[threadIdx.x], red[threadIdx.x + s]);
        __syncthreads();
    }
    m = red[0]; __syncthreads();
    float ss = 0.f;
    for (int v = threadIdx.x; v < HV; v += 256)
        ss += __expf(EmT[(size_t)v * HK + k] - m);
    red[threadIdx.x] = ss; __syncthreads();
    for (int s = 128; s > 0; s >>= 1) {
        if (threadIdx.x < s) red[threadIdx.x] += red[threadIdx.x + s];
        __syncthreads();
    }
    if (threadIdx.x == 0) lse_k[k] = m + __logf(red[0]);
}

// ---------------------------------------------------------------------------
// Kernel 4: fused tran-GEMM (WMMA) + row-lse folding + alpha scan.
// One block per batch element n. 512 threads = 16 wave32.
// ---------------------------------------------------------------------------
__global__ __launch_bounds__(512) void k_scan(const int* __restrict__ tokens,
                                              const float* __restrict__ emb_w,
                                              const float* __restrict__ start_w,
                                              const float* __restrict__ start_b,
                                              const unsigned short* __restrict__ trans_bf,
                                              const float* __restrict__ EmT,
                                              const float* __restrict__ lse_k,
                                              float* __restrict__ nres) {
    extern __shared__ unsigned char smem[];
    unsigned short* tran_s = (unsigned short*)smem;          // 16*4096 bf16 = 128 KB
    unsigned short* x_s    = tran_s + TB * HE;               // 16*128  bf16 =   4 KB
    float* rowlse = (float*)(x_s + TB * DPAD);               // 16*64 f32
    float* abuf0  = rowlse + TB * HK;                        // 64 (ping)
    float* abuf1  = abuf0 + HK;                              // 64 (pong)
    float* pm     = abuf1 + HK;                              // 8*64 partial max
    float* ps     = pm + SCAN_G * HK;                        // 8*64 partial sum

    const int n    = blockIdx.x;
    const int tid  = threadIdx.x;
    const int lane = tid & 31;
    const int hi   = lane >> 4;
    const int mrow = lane & 15;
    const int wave = tid >> 5;                               // 0..15

    // alpha0 = log_softmax(start_w + start_b)  -> abuf0
    if (tid < HK) abuf0[tid] = start_w[tid] + start_b[tid];
    __syncthreads();
    if (tid < HK) {
        float m = NEGINF;
        for (int i = 0; i < HK; ++i) m = fmaxf(m, abuf0[i]);
        float s = 0.f;
        for (int i = 0; i < HK; ++i) s += __expf(abuf0[i] - m);
        abuf1[tid] = abuf0[tid] - (m + __logf(s));
    }
    __syncthreads();
    if (tid < HK) abuf0[tid] = abuf1[tid];
    __syncthreads();

    int cur = 0;                                             // alpha lives in abuf[cur]

    for (int c = 0; c < 32; ++c) {                   // 32 chunks cover t=0..510
        const int t0 = c * TB;
        const int nt = (511 - t0 < TB) ? (511 - t0) : TB;

        // ---- stage x[n, t0..t0+nt) as bf16, zero-padded to DPAD ----
        for (int idx = tid; idx < TB * DPAD; idx += 512) {
            int r = idx >> 7, d = idx & 127;
            float v = 0.f;
            if (r < nt && d < HD) {
                int tok = tokens[n * HT + t0 + r];
                v = emb_w[(size_t)tok * HD + d];
            }
            x_s[idx] = f2bf(v);
        }
        __syncthreads();

        // ---- WMMA: tran[t, e] = sum_d x[t,d] * trans_w[e,d] ----
        V16 afr[4];
        const unsigned short* xr = x_s + mrow * DPAD;
#pragma unroll
        for (int kk = 0; kk < 4; ++kk) {
            int b0 = kk * 32 + hi * 8;
            afr[kk].q[0] = *(const uint4*)(xr + b0);
            afr[kk].q[1] = *(const uint4*)(xr + b0 + 16);
        }
        const int et0 = wave * 16;                   // 256 e-tiles / 16 waves
        for (int et = et0; et < et0 + 16; ++et) {
            const int erow = et * 16 + mrow;         // this lane's trans_w row
            const unsigned short* brow = trans_bf + (size_t)erow * DPAD + hi * 8;
            // prefetch next tile's row for this lane (global_prefetch_b8)
            __builtin_prefetch(trans_bf + (size_t)((et + 1) * 16 + mrow) * DPAD, 0, 1);
            v8f acc = {0.f,0.f,0.f,0.f,0.f,0.f,0.f,0.f};
#pragma unroll
            for (int kk = 0; kk < 4; ++kk) {
                V16 bu;
                const uint4* pb = (const uint4*)(brow + kk * 32);
                bu.q[0] = pb[0];
                bu.q[1] = pb[2];
                acc = __builtin_amdgcn_wmma_f32_16x16x32_bf16(
                    false, afr[kk].v, false, bu.v, (short)0, acc, false, false);
            }
#pragma unroll
            for (int r = 0; r < 8; ++r) {
                int mm = r + hi * 8;                 // timestep row
                tran_s[mm * HE + et * 16 + mrow] = f2bf(acc[r]);
            }
        }
        __syncthreads();

        // ---- row-wise lse over j (folds tran log_softmax into alpha) ----
        for (int row = tid; row < TB * HK; row += 512) {
            int t = row >> 6, i = row & 63;
            if (t < nt) {
                const unsigned short* tr = tran_s + t * HE + i * HK;
                float m = NEGINF;
                for (int j = 0; j < HK; ++j) m = fmaxf(m, bf2f(tr[j]));
                float s = 0.f;
                for (int j = 0; j < HK; ++j) s += __expf(bf2f(tr[j]) - m);
                rowlse[row] = m + __logf(s);
            }
        }
        __syncthreads();

        // ---- alpha scan: 8-way-parallel i-reduction, ping-pong buffers ----
        const int j = tid & 63;
        const int g = tid >> 6;                      // 0..7, covers i = g*8..g*8+7
        for (int tt = 0; tt < nt; ++tt) {
            const float* ac = cur ? abuf1 : abuf0;
            float* an       = cur ? abuf0 : abuf1;
            const unsigned short* tr = tran_s + tt * HE;
            const float* rl = rowlse + tt * HK;

            float vv[8];
            float m = NEGINF;
#pragma unroll
            for (int ii = 0; ii < 8; ++ii) {
                const int i = g * 8 + ii;
                vv[ii] = ac[i] - rl[i] + bf2f(tr[i * HK + j]);
                m = fmaxf(m, vv[ii]);
            }
            float s = 0.f;
#pragma unroll
            for (int ii = 0; ii < 8; ++ii) s += __expf(vv[ii] - m);
            pm[g * HK + j] = m;
            ps[g * HK + j] = s;
            __syncthreads();

            if (tid < HK) {
                float M = NEGINF;
#pragma unroll
                for (int g2 = 0; g2 < SCAN_G; ++g2) M = fmaxf(M, pm[g2 * HK + tid]);
                float S = 0.f;
#pragma unroll
                for (int g2 = 0; g2 < SCAN_G; ++g2)
                    S += ps[g2 * HK + tid] * __expf(pm[g2 * HK + tid] - M);
                const int tok = tokens[n * HT + t0 + tt + 1];
                const float emit = EmT[(size_t)tok * HK + tid] - lse_k[tid];
                an[tid] = M + __logf(S) + emit;
            }
            __syncthreads();
            cur ^= 1;
        }
        __syncthreads();
    }

    if (tid == 0) {
        const float* ac = cur ? abuf1 : abuf0;
        float m = NEGINF;
        for (int i = 0; i < HK; ++i) m = fmaxf(m, ac[i]);
        float s = 0.f;
        for (int i = 0; i < HK; ++i) s += __expf(ac[i] - m);
        nres[n] = m + __logf(s);
    }
}

// ---------------------------------------------------------------------------
// Kernel 5: out = -mean(nres)
// ---------------------------------------------------------------------------
__global__ void k_final(const float* __restrict__ nres, float* __restrict__ out) {
    if (threadIdx.x == 0) {
        float s = 0.f;
        for (int i = 0; i < HN; ++i) s += nres[i];
        out[0] = -s / (float)HN;
    }
}

// ---------------------------------------------------------------------------
extern "C" void kernel_launch(void* const* d_in, const int* in_sizes, int n_in,
                              void* d_out, int out_size, void* d_ws, size_t ws_size,
                              hipStream_t stream) {
    (void)in_sizes; (void)n_in; (void)out_size; (void)ws_size;
    const int*   tokens  = (const int*)d_in[0];
    const float* emb_w   = (const float*)d_in[1];
    const float* start_w = (const float*)d_in[2];
    const float* start_b = (const float*)d_in[3];
    const float* embc    = (const float*)d_in[4];
    const float* vocab   = (const float*)d_in[5];
    const float* trans_w = (const float*)d_in[6];

    unsigned char* ws = (unsigned char*)d_ws;
    unsigned short* trans_bf = (unsigned short*)ws;                    // 1,048,576 B
    float* EmT   = (float*)(ws + (size_t)HE * DPAD * 2);               // 8,192,000 B
    float* lse_k = (float*)(ws + (size_t)HE * DPAD * 2 + (size_t)HV * HK * 4);
    float* nres  = lse_k + HK;

    k_prep_trans<<<(HE * DPAD) / 256, 256, 0, stream>>>(trans_w, trans_bf);
    k_em<<<HV / 128, 256, 0, stream>>>(embc, vocab, EmT);
    k_lse<<<HK, 256, 0, stream>>>(EmT, lse_k);

    const size_t shmem = (size_t)TB * HE * 2      // tran chunk (bf16)
                       + (size_t)TB * DPAD * 2    // x stage (bf16)
                       + (size_t)TB * HK * 4      // row lse
                       + 2 * HK * 4               // alpha ping-pong
                       + 2 * SCAN_G * HK * 4;     // partial max/sum
    k_scan<<<HN, 512, shmem, stream>>>(tokens, emb_w, start_w, start_b,
                                       trans_bf, EmT, lse_k, nres);
    k_final<<<1, 32, 0, stream>>>(nres, (float*)d_out);
}
